// MLA_50379966382638
// MI455X (gfx1250) — matrix operand order repaired and checked by reading, not compiled, not run on hardware
//
#include <hip/hip_runtime.h>
#include <hip/hip_bf16.h>
#include <stdint.h>

// ---------------------------------------------------------------------------
// MLA forward for MI455X (gfx1250, wave32, WMMA + TDM + async-LDS).
// fp32->bf16 convert, double-buffered bf16 NT GEMMs (WMMA 16x16x32, TDM for
// weight tiles, async-to-LDS via inline asm for activation tiles), flash
// attention (WMMA QK^T + PV, online softmax), final GEMM + bias in fp32.
// ---------------------------------------------------------------------------

typedef __attribute__((ext_vector_type(16))) __bf16 bf16x16;
typedef __attribute__((ext_vector_type(8)))  float  f32x8;

#if __has_builtin(__builtin_amdgcn_tensor_load_to_lds)
#define HAS_TDM 1
#else
#define HAS_TDM 0
#endif

union Frag16 {
    bf16x16 v;
    uint4   q[2];
    unsigned short u[16];
};

__device__ __forceinline__ unsigned short f32_to_bf16(float f) {
    uint32_t u = __float_as_uint(f);
    uint32_t r = u + 0x7FFFu + ((u >> 16) & 1u);
    return (unsigned short)(r >> 16);
}

// --- 16B global->LDS async copy (VGLOBAL async encoding, GV mode) ----------
// VDST carries the LDS byte address; VADDR the 64-bit global address.
__device__ __forceinline__ void cp16_g2l(const unsigned short* g, unsigned short* l) {
    const uint64_t ga = (uint64_t)(uintptr_t)g;
    const uint32_t la = (uint32_t)(uintptr_t)l;   // generic LDS addr -> [31:0]
    asm volatile("global_load_async_to_lds_b128 %0, %1, off"
                 :: "v"(la), "v"(ga)
                 : "memory");
}

__device__ __forceinline__ void wait_async0() {
    asm volatile("s_wait_asynccnt 0x0" ::: "memory");
}

__device__ __forceinline__ void wait_tensor0() {
#if HAS_TDM
#if __has_builtin(__builtin_amdgcn_s_wait_tensorcnt)
    __builtin_amdgcn_s_wait_tensorcnt(0);
#else
    asm volatile("s_wait_tensorcnt 0x0" ::: "memory");
#endif
#endif
}

#if HAS_TDM
typedef __attribute__((ext_vector_type(4))) unsigned int u32x4_t;
typedef __attribute__((ext_vector_type(8))) int          i32x8_t;
typedef __attribute__((ext_vector_type(4))) int          i32x4_t;

// 2-D TDM load: tileK x tileRows bf16 tile, row stride rowStrideElems, into
// LDS with hardware padding (padded row = tileK elems + padAmount DWORDs).
// Descriptor packing per CDNA5 ISA.
__device__ __forceinline__ void tdm_load_2d(
    const unsigned short* gsrc, unsigned short* ldst,
    int tileK, int tileRows, long rowStrideElems,
    long tensorD0, long tensorD1,
    int padIntervalCode, int padAmountCode)
{
    const uint64_t ga = (uint64_t)(uintptr_t)gsrc;
    const uint32_t la = (uint32_t)(uintptr_t)ldst;   // generic LDS addr -> [31:0]

    u32x4_t g0 = { 1u,                               // count=1, user descriptor
                   la,                               // lds_addr
                   (uint32_t)ga,                     // global_addr[31:0]
                   (uint32_t)((ga >> 32) & 0x01FFFFFFu) | (2u << 30) }; // type=2

    const uint32_t flags = (1u << 16)                               // data_size=2B
                         | (padAmountCode ? (1u << 20) : 0u)        // pad_enable
                         | ((uint32_t)padIntervalCode << 22)
                         | ((uint32_t)padAmountCode << 25);
    const uint64_t d0 = (uint64_t)tensorD0, d1 = (uint64_t)tensorD1;
    const uint64_t s0 = (uint64_t)rowStrideElems, s1 = 0;

    i32x8_t g1 = { (int)flags,
                   (int)((d0 & 0xFFFFu) << 16),                                   // dim0[15:0]
                   (int)((uint32_t)(d0 >> 16) | (uint32_t)((d1 & 0xFFFFu) << 16)),// dim0[31:16],dim1[15:0]
                   (int)((uint32_t)((d1 >> 16) & 0xFFFFu) | ((uint32_t)tileK << 16)), // dim1[31:16],tile0
                   (int)(uint32_t)tileRows,                                       // tile1 (tile2=0)
                   (int)(uint32_t)s0,                                             // stride0[31:0]
                   (int)((uint32_t)((s0 >> 32) & 0xFFFFu) | (uint32_t)((s1 & 0xFFFFu) << 16)),
                   (int)(uint32_t)(s1 >> 16) };
    i32x4_t z4 = (i32x4_t)0;
#if __has_include(<hip/amd_detail/amd_gfx1250_TDM.h>)
    __builtin_amdgcn_tensor_load_to_lds(g0, g1, z4, z4, (i32x8_t)0, 0);   // clang-23 arity
#else
    __builtin_amdgcn_tensor_load_to_lds(g0, g1, z4, z4, 0);               // ROCm 7.2 arity
#endif
}
#endif  // HAS_TDM

// ---------------------------------------------------------------------------
// 1) fp32 -> bf16 packed conversion
// ---------------------------------------------------------------------------
__global__ void cvt_f32_bf16_kernel(const float* __restrict__ src,
                                    uint2* __restrict__ dst, long n4) {
    long i = (long)blockIdx.x * blockDim.x + threadIdx.x;
    long stride = (long)gridDim.x * blockDim.x;
    for (; i < n4; i += stride) {
        float4 f = ((const float4*)src)[i];
        uint2 o;
        o.x = ((uint32_t)f32_to_bf16(f.y) << 16) | f32_to_bf16(f.x);
        o.y = ((uint32_t)f32_to_bf16(f.w) << 16) | f32_to_bf16(f.z);
        dst[i] = o;
    }
}

// ---------------------------------------------------------------------------
// 2) NT GEMM:  C[M,N] = A[M,K] @ B[N,K]^T  (bf16 in, fp32 acc)
//    Double buffered; A tile via async-to-LDS, B tile via one TDM DMA.
// ---------------------------------------------------------------------------
template <bool OUT_BF16, bool HAS_BIAS>
__global__ __launch_bounds__(256) void gemm_nt_kernel(
    const unsigned short* __restrict__ A,   // [M,K]
    const unsigned short* __restrict__ B,   // [N,K]
    void* __restrict__ C,                   // [M,N]
    const float* __restrict__ bias,         // [N] or nullptr
    int M, int N, int K)
{
    constexpr int BM = 128, BN = 128, BK = 32;
    constexpr int LDT = BK + 8;             // 40 elems = 80B: 16B aligned, conflict-free
    __shared__ unsigned short As[2][BM * LDT];
    __shared__ unsigned short Bs[2][BN * LDT];

    const int tid  = threadIdx.x;
    const int lane = tid & 31;
    const int wave = tid >> 5;
    const int wm   = wave >> 1;             // 0..3
    const int wn   = wave & 1;              // 0..1

    const int blockM = blockIdx.y * BM;
    const int blockN = blockIdx.x * BN;

    const int ldRow  = tid >> 1;            // 0..127
    const int ldHalf = (tid & 1) * 16;

    const int mLane = lane & 15;
    const int kGrpA = (lane >> 4) * 8;
    const int kGrpB = (lane >> 4) * 8;
    const int rGrp  = (lane >> 4) * 8;

    f32x8 acc[2][4];
#pragma unroll
    for (int i = 0; i < 2; ++i)
#pragma unroll
        for (int j = 0; j < 4; ++j)
#pragma unroll
            for (int e = 0; e < 8; ++e) acc[i][j][e] = 0.0f;

    const unsigned short* ag0 = A + (size_t)(blockM + ldRow) * K + ldHalf;
    const unsigned short* bg0 = B + (size_t)(blockN + ldRow) * K + ldHalf;
    const unsigned short* bt0 = B + (size_t)blockN * K;       // TDM tile origin

    auto stage = [&](int k0, int buf) {
        const unsigned short* ag = ag0 + k0;
        cp16_g2l(ag,     &As[buf][ldRow * LDT + ldHalf]);
        cp16_g2l(ag + 8, &As[buf][ldRow * LDT + ldHalf + 8]);
#if HAS_TDM
        if (wave == 0)
            // 32x128 bf16 tile; pad_interval: 16 DWORDs (=64B row),
            // pad_amount: 4 DWORDs (=8 elems) -> padded row = LDT elems.
            tdm_load_2d(bt0 + k0, &Bs[buf][0], BK, BN, K, K, BN, 3, 3);
#else
        const unsigned short* bg = bg0 + k0;
        cp16_g2l(bg,     &Bs[buf][ldRow * LDT + ldHalf]);
        cp16_g2l(bg + 8, &Bs[buf][ldRow * LDT + ldHalf + 8]);
#endif
        if (k0 + BK < K) __builtin_prefetch(ag + BK, 0, 1);   // L2 prefetch
    };

    stage(0, 0);
    wait_async0();
    if (wave == 0) wait_tensor0();
    __syncthreads();

    int buf = 0;
    for (int k0 = 0; k0 < K; k0 += BK) {
        if (k0 + BK < K) stage(k0 + BK, buf ^ 1);

        Frag16 bf[4];
#pragma unroll
        for (int tn = 0; tn < 4; ++tn) {
            const int row = wn * 64 + tn * 16 + mLane;
            bf[tn].q[0] = *(const uint4*)&Bs[buf][row * LDT + kGrpB];
            bf[tn].q[1] = *(const uint4*)&Bs[buf][row * LDT + kGrpB + 16];
        }
#pragma unroll
        for (int tm = 0; tm < 2; ++tm) {
            Frag16 af;
            const int row = wm * 32 + tm * 16 + mLane;
            af.q[0] = *(const uint4*)&As[buf][row * LDT + kGrpA];
            af.q[1] = *(const uint4*)&As[buf][row * LDT + 16 + kGrpA];
#pragma unroll
            for (int tn = 0; tn < 4; ++tn)
                acc[tm][tn] = __builtin_amdgcn_wmma_f32_16x16x32_bf16(
                    false, af.v, false, bf[tn].v, (short)0, acc[tm][tn], false, false);
        }

        wait_async0();                 // next tile fully in LDS
        if (wave == 0) wait_tensor0();
        __syncthreads();
        buf ^= 1;
    }

#pragma unroll
    for (int tm = 0; tm < 2; ++tm)
#pragma unroll
        for (int tn = 0; tn < 4; ++tn) {
            const int col = blockN + wn * 64 + tn * 16 + mLane;
            const float bv = HAS_BIAS ? bias[col] : 0.0f;
#pragma unroll
            for (int r = 0; r < 8; ++r) {
                const int row = blockM + wm * 32 + tm * 16 + rGrp + r;
                const float val = acc[tm][tn][r] + bv;
                if (OUT_BF16)
                    ((unsigned short*)C)[(size_t)row * N + col] = f32_to_bf16(val);
                else
                    ((float*)C)[(size_t)row * N + col] = val;
            }
        }
}

// ---------------------------------------------------------------------------
// 3) Flash attention (causal), latent head space.
//    128 q-rows per block (8 waves x 16 rows). 64-key blocks; K tile async
//    into LDS, V staged transposed, P via LDS for layout conversion.
// ---------------------------------------------------------------------------
__global__ __launch_bounds__(256) void mla_attn_kernel(
    const unsigned short* __restrict__ qlat,   // [B*S, 1024]
    const unsigned short* __restrict__ kvlat,  // [B*S, 1024]
    const unsigned short* __restrict__ vfull,  // [B*S, 2048]
    unsigned short* __restrict__ attno)        // [B*S, 2048]
{
    constexpr int SEQ = 2048, Rr = 1024, HV = 2048, VD = 128;
    constexpr float SC = 0.047245559126f;      // 1/sqrt(448)
    constexpr int PLD = 72, VLD = 72, KLD = 72;

    __shared__ unsigned short Pl[8 * 16 * PLD];
    __shared__ unsigned short Vt[VD * VLD];
    __shared__ unsigned short Ks[64 * KLD];

    const int qt = blockIdx.x;
    const int h  = blockIdx.y;
    const int b  = blockIdx.z;
    const int qbase = qt * 128;

    const int tid  = threadIdx.x;
    const int lane = tid & 31;
    const int wave = tid >> 5;
    const int mLane = lane & 15;
    const int kGrpA = (lane >> 4) * 8;
    const int kGrpB = (lane >> 4) * 8;
    const int rGrp  = (lane >> 4) * 8;

    const int qrow0 = qbase + wave * 16;
    const size_t grow = (size_t)(b * SEQ + qrow0);

    Frag16 qf[2];
#pragma unroll
    for (int s = 0; s < 2; ++s) {
        const unsigned short* qp = qlat + (grow + mLane) * Rr + h * 64 + s * 32;
        qf[s].q[0] = *(const uint4*)(qp + kGrpA);
        qf[s].q[1] = *(const uint4*)(qp + 16 + kGrpA);
    }

    f32x8 oacc[8];
#pragma unroll
    for (int t = 0; t < 8; ++t)
#pragma unroll
        for (int e = 0; e < 8; ++e) oacc[t][e] = 0.0f;

    float mrun[8], lrun[8];
#pragma unroll
    for (int r = 0; r < 8; ++r) { mrun[r] = -1e30f; lrun[r] = 0.0f; }

    const int nkb = qt * 2 + 2;
    for (int kb = 0; kb < nkb; ++kb) {
        const int kbase = kb * 64;

        __syncthreads();        // previous-iteration LDS reads complete
        {   // K tile (64 keys x 64 hd) via async-to-LDS; 4 threads/row
            const int krow = tid >> 2;
            const int kch  = (tid & 3) * 16;
            const unsigned short* kp =
                kvlat + (size_t)(b * SEQ + kbase + krow) * Rr + h * 64 + kch;
            cp16_g2l(kp,     &Ks[krow * KLD + kch]);
            cp16_g2l(kp + 8, &Ks[krow * KLD + kch + 8]);
        }
        {   // V tile transposed (vd x key); lane-granular b16 writes
            const int key = tid >> 2;
            const int ch  = (tid & 3) * 32;
            const unsigned short* vp =
                vfull + (size_t)(b * SEQ + kbase + key) * HV + h * VD + ch;
#pragma unroll
            for (int c4 = 0; c4 < 4; ++c4) {
                uint4 vv = *(const uint4*)(vp + c4 * 8);
                const unsigned short* pe = (const unsigned short*)&vv;
#pragma unroll
                for (int e = 0; e < 8; ++e)
                    Vt[(ch + c4 * 8 + e) * VLD + key] = pe[e];
            }
        }
        wait_async0();
        __syncthreads();

        // ---- S = Q K^T -------------------------------------------------
        f32x8 sacc[4];
#pragma unroll
        for (int t = 0; t < 4; ++t)
#pragma unroll
            for (int e = 0; e < 8; ++e) sacc[t][e] = 0.0f;

#pragma unroll
        for (int s = 0; s < 2; ++s) {
#pragma unroll
            for (int tn = 0; tn < 4; ++tn) {
                Frag16 kf;
                const unsigned short* kp =
                    &Ks[(tn * 16 + mLane) * KLD + s * 32 + kGrpB];
                kf.q[0] = *(const uint4*)(kp);
                kf.q[1] = *(const uint4*)(kp + 16);
                sacc[tn] = __builtin_amdgcn_wmma_f32_16x16x32_bf16(
                    false, qf[s].v, false, kf.v, (short)0, sacc[tn], false, false);
            }
        }

        // ---- scale + causal mask + row max -----------------------------
        float rmax[8];
#pragma unroll
        for (int r = 0; r < 8; ++r) rmax[r] = -1e30f;
#pragma unroll
        for (int tn = 0; tn < 4; ++tn) {
            const int key = kbase + tn * 16 + mLane;
#pragma unroll
            for (int r = 0; r < 8; ++r) {
                const int qrow = qrow0 + rGrp + r;
                float v = sacc[tn][r] * SC;
                v = (key <= qrow) ? v : -1e30f;
                sacc[tn][r] = v;
                rmax[r] = fmaxf(rmax[r], v);
            }
        }
#pragma unroll
        for (int off = 1; off < 16; off <<= 1)
#pragma unroll
            for (int r = 0; r < 8; ++r)
                rmax[r] = fmaxf(rmax[r], __shfl_xor(rmax[r], off, 32));

        float alpha[8], rsum[8];
#pragma unroll
        for (int r = 0; r < 8; ++r) {
            const float mnew = fmaxf(mrun[r], rmax[r]);
            alpha[r] = __expf(mrun[r] - mnew);
            mrun[r] = mnew;
            rsum[r] = 0.0f;
        }

#pragma unroll
        for (int tn = 0; tn < 4; ++tn)
#pragma unroll
            for (int r = 0; r < 8; ++r) {
                const float p = __expf(sacc[tn][r] - mrun[r]);
                rsum[r] += p;
                Pl[(wave * 16 + rGrp + r) * PLD + tn * 16 + mLane] = f32_to_bf16(p);
            }
#pragma unroll
        for (int off = 1; off < 16; off <<= 1)
#pragma unroll
            for (int r = 0; r < 8; ++r)
                rsum[r] += __shfl_xor(rsum[r], off, 32);
#pragma unroll
        for (int r = 0; r < 8; ++r) lrun[r] = lrun[r] * alpha[r] + rsum[r];

#pragma unroll
        for (int t = 0; t < 8; ++t)
#pragma unroll
            for (int r = 0; r < 8; ++r) oacc[t][r] *= alpha[r];

        // ---- O += P @ V ------------------------------------------------
#pragma unroll
        for (int s2 = 0; s2 < 2; ++s2) {
            Frag16 pf;
            const unsigned short* pp = &Pl[(wave * 16 + mLane) * PLD + s2 * 32];
            pf.q[0] = *(const uint4*)(pp + kGrpA);
            pf.q[1] = *(const uint4*)(pp + 16 + kGrpA);
#pragma unroll
            for (int tn = 0; tn < 8; ++tn) {
                Frag16 vf;
                const unsigned short* vv =
                    &Vt[(tn * 16 + mLane) * VLD + s2 * 32 + kGrpB];
                vf.q[0] = *(const uint4*)(vv);
                vf.q[1] = *(const uint4*)(vv + 16);
                oacc[tn] = __builtin_amdgcn_wmma_f32_16x16x32_bf16(
                    false, pf.v, false, vf.v, (short)0, oacc[tn], false, false);
            }
        }
    }

#pragma unroll
    for (int tn = 0; tn < 8; ++tn)
#pragma unroll
        for (int r = 0; r < 8; ++r) {
            const float o = oacc[tn][r] / lrun[r];
            const int row = qrow0 + rGrp + r;
            attno[(size_t)(b * SEQ + row) * HV + h * VD + tn * 16 + mLane] =
                f32_to_bf16(o);
        }
}

// ---------------------------------------------------------------------------
// Host-side orchestration
// ---------------------------------------------------------------------------
extern "C" void kernel_launch(void* const* d_in, const int* in_sizes, int n_in,
                              void* d_out, int out_size, void* d_ws, size_t ws_size,
                              hipStream_t stream) {
    (void)in_sizes; (void)n_in; (void)out_size; (void)ws_size;

    const float* x   = (const float*)d_in[0];
    const float* Wq  = (const float*)d_in[1];
    const float* Wkv = (const float*)d_in[2];
    const float* Wvb = (const float*)d_in[3];
    const float* Wo  = (const float*)d_in[4];
    const float* bo  = (const float*)d_in[5];

    constexpr long BS = 4096, D = 7168, R = 1024, HV = 2048;

    unsigned short* ws    = (unsigned short*)d_ws;
    unsigned short* xb    = ws;
    unsigned short* wqb   = xb    + BS * D;
    unsigned short* wkvb  = wqb   + R * D;
    unsigned short* wvbb  = wkvb  + R * D;
    unsigned short* wob   = wvbb  + HV * R;
    unsigned short* ql    = wob   + D * HV;
    unsigned short* kvl   = ql    + BS * R;
    unsigned short* vfull = kvl   + BS * R;
    unsigned short* attno = vfull + BS * HV;

    auto cvt = [&](const float* s, unsigned short* dvec, long n) {
        const long n4 = n / 4;
        long nb = (n4 + 255) / 256;
        if (nb > 8192) nb = 8192;
        cvt_f32_bf16_kernel<<<dim3((unsigned)nb), dim3(256), 0, stream>>>(
            s, (uint2*)dvec, n4);
    };
    cvt(x,   xb,   BS * D);
    cvt(Wq,  wqb,  R * D);
    cvt(Wkv, wkvb, R * D);
    cvt(Wvb, wvbb, HV * R);
    cvt(Wo,  wob,  D * HV);

    const dim3 blk(256);
    gemm_nt_kernel<true, false><<<dim3(R / 128, BS / 128), blk, 0, stream>>>(
        xb, wqb, ql, nullptr, (int)BS, (int)R, (int)D);
    gemm_nt_kernel<true, false><<<dim3(R / 128, BS / 128), blk, 0, stream>>>(
        xb, wkvb, kvl, nullptr, (int)BS, (int)R, (int)D);
    gemm_nt_kernel<true, false><<<dim3(HV / 128, BS / 128), blk, 0, stream>>>(
        kvl, wvbb, vfull, nullptr, (int)BS, (int)HV, (int)R);
    mla_attn_kernel<<<dim3(16, 16, 2), blk, 0, stream>>>(ql, kvl, vfull, attno);
    gemm_nt_kernel<false, true><<<dim3(D / 128, BS / 128), blk, 0, stream>>>(
        attno, wob, (float*)d_out, bo, (int)BS, (int)D, (int)HV);
}
